// CMPNNEncoder_40699110097566
// MI455X (gfx1250) — compile-verified, matrix operand order checked by reading, not run on hardware
//
#include <hip/hip_runtime.h>
#include <hip/hip_bf16.h>

#define NN 100000
#define NE 400000
#define HID 256
#define ATOM_F 133
#define BOND_F 147
#define KP_IN 160   // 133/147 padded up to multiple of 32
#define LDS_PAD 8   // bf16 elements of row padding in LDS (16B -> +4 banks/row)

typedef __attribute__((ext_vector_type(16))) __bf16 v16bf;
typedef __attribute__((ext_vector_type(8)))  __bf16 v8bf;
typedef __attribute__((ext_vector_type(4)))  __bf16 v4bf;
typedef __attribute__((ext_vector_type(8)))  float  v8f;

static __device__ __forceinline__ __bf16 f2bf(float f) { return (__bf16)f; }

#define SHUF16(lo, hi) __builtin_shufflevector(lo, hi, 0,1,2,3,4,5,6,7,8,9,10,11,12,13,14,15)

#define KSTEP(bq0, bq1, k0)                                                    \
    {                                                                          \
        v8bf a0l = *(const v8bf*)(ap0 + (k0));                                 \
        v8bf a0h = *(const v8bf*)(ap0 + (k0) + 16);                            \
        v8bf a1l = *(const v8bf*)(ap1 + (k0));                                 \
        v8bf a1h = *(const v8bf*)(ap1 + (k0) + 16);                            \
        v8bf b0l = *(const v8bf*)((bq0) + (k0));                               \
        v8bf b0h = *(const v8bf*)((bq0) + (k0) + 16);                          \
        v8bf b1l = *(const v8bf*)((bq1) + (k0));                               \
        v8bf b1h = *(const v8bf*)((bq1) + (k0) + 16);                          \
        v16bf a0 = SHUF16(a0l, a0h);                                           \
        v16bf a1 = SHUF16(a1l, a1h);                                           \
        v16bf b0 = SHUF16(b0l, b0h);                                           \
        v16bf b1 = SHUF16(b1l, b1h);                                           \
        cc[0][0] = __builtin_amdgcn_wmma_f32_16x16x32_bf16(false, a0, false, b0, (short)0, cc[0][0], false, false); \
        cc[0][1] = __builtin_amdgcn_wmma_f32_16x16x32_bf16(false, a0, false, b1, (short)0, cc[0][1], false, false); \
        cc[1][0] = __builtin_amdgcn_wmma_f32_16x16x32_bf16(false, a1, false, b0, (short)0, cc[1][0], false, false); \
        cc[1][1] = __builtin_amdgcn_wmma_f32_16x16x32_bf16(false, a1, false, b1, (short)0, cc[1][1], false, false); \
    }

// ---------------------------------------------------------------------------
// bf16 WMMA GEMM, 2x2 register blocking: each wave computes a 32x32 output
// tile of out[M,256] = post( A[M,Kp] @ Bt[256,Kp]^T ). 8 waves/block cover all
// 8 N-bands of one 32-row stripe. When use_lds != 0, the whole Bt operand is
// staged into LDS once per block via global_load_async_to_lds_b128 (ASYNCcnt),
// then fragments come from ds_load_b128.
// 16-bit operand fragment layout (wave32): lane L -> row L&15, half h = L>>4,
// holds K in [k0+8h, k0+8h+8) U [k0+16+8h, k0+16+8h+8).
// ---------------------------------------------------------------------------
__global__ __launch_bounds__(256) void k_gemm_bf16(
    float* __restrict__ out, const __bf16* __restrict__ A,
    const __bf16* __restrict__ Bt, const float* __restrict__ addsrc,
    const float* __restrict__ bias, long M, int Kp, int relu, int use_lds)
{
    extern __shared__ __bf16 ldsB[];   // [256][Kp + LDS_PAD] when use_lds

    const int tid  = threadIdx.x;
    const int lane = tid & 31;
    const int wave = tid >> 5;
    const long m0  = (long)blockIdx.x * 32;
    const int  n0  = wave * 32;
    if (m0 >= M) return;

    const int half = lane >> 4;
    const int l15  = lane & 15;
    const int ldsKp = Kp + LDS_PAD;

    if (use_lds) {
        // async-stage Bt (256*Kp bf16) into padded LDS rows, 16B granules
        const unsigned ldsbase = (unsigned)(size_t)(void*)ldsB;
        const int nchunks = (HID * Kp) / 8;          // 16B granules
        for (int c = tid; c < nchunks; c += 256) {
            int el  = c * 8;
            int row = el / Kp;
            int col = el - row * Kp;
            unsigned loff = ldsbase + (unsigned)(row * ldsKp + col) * 2u;
            const void* gaddr = (const void*)(Bt + (size_t)el);
            asm volatile("global_load_async_to_lds_b128 %0, %1, off"
                         :: "v"(loff), "v"(gaddr) : "memory");
        }
        asm volatile("s_wait_asynccnt 0x0" ::: "memory");
        __syncthreads();
    }

    const __bf16* ap0 = A + (m0 + l15) * (long)Kp + half * 8;
    const __bf16* ap1 = ap0 + 16 * (long)Kp;

    v8f cc[2][2] = {};
    if (use_lds) {
        const __bf16* bq0 = ldsB + (size_t)(n0 + l15) * ldsKp + half * 8;
        const __bf16* bq1 = bq0 + 16 * ldsKp;
        for (int k0 = 0; k0 < Kp; k0 += 32) KSTEP(bq0, bq1, k0)
    } else {
        const __bf16* bq0 = Bt + (size_t)(n0 + l15) * Kp + half * 8;
        const __bf16* bq1 = bq0 + 16 * Kp;
        for (int k0 = 0; k0 < Kp; k0 += 32) KSTEP(bq0, bq1, k0)
    }

    // D layout: VGPR r -> row sub_m0 + 8*half + r, col sub_n0 + (lane&15)
#pragma unroll
    for (int i = 0; i < 2; ++i) {
#pragma unroll
        for (int j = 0; j < 2; ++j) {
            const int  n  = n0 + j * 16 + l15;
            const long rb = m0 + i * 16 + half * 8;
#pragma unroll
            for (int r = 0; r < 8; ++r) {
                const long m = rb + r;
                float v = cc[i][j][r];
                if (addsrc) v += addsrc[m * HID + n];
                if (bias)   v += bias[n];
                if (relu)   v = v > 0.f ? v : 0.f;
                out[m * HID + n] = v;
            }
        }
    }
}

// Zero-padded f32 -> bf16 row copy (node/edge features: src_cols -> dst_cols)
__global__ void k_pad_bf16(__bf16* __restrict__ dst, const float* __restrict__ src,
                           long rows, int sc, int dc)
{
    long t = (long)blockIdx.x * blockDim.x + threadIdx.x;
    if (t >= rows * dc) return;
    long r = t / dc; int c = (int)(t - r * dc);
    dst[t] = f2bf(c < sc ? src[r * sc + c] : 0.f);
}

// Weight [K,256] f32 -> transposed padded [256,Kp] bf16
__global__ void k_wt_bf16(__bf16* __restrict__ dst, const float* __restrict__ src,
                          int K, int Kp)
{
    int t = blockIdx.x * blockDim.x + threadIdx.x;
    if (t >= HID * Kp) return;
    int n = t / Kp, k = t - n * Kp;
    dst[t] = f2bf(k < K ? src[(long)k * HID + n] : 0.f);
}

__global__ void k_copy_f32(float* __restrict__ d, const float* __restrict__ s, long n)
{
    long t = (long)blockIdx.x * blockDim.x + threadIdx.x;
    if (t < n) d[t] = s[t];
}

__global__ void k_fill0_f32(float* __restrict__ d, long n)
{
    long t = (long)blockIdx.x * blockDim.x + threadIdx.x;
    if (t < n) d[t] = 0.f;
}

// dstbuf[dst[e]][:] += h_edge[e][:]   (f32 atomics, 4 cols per thread)
__global__ void k_scatter_add(float* __restrict__ dstbuf, const float* __restrict__ he,
                              const int* __restrict__ dsti, long E)
{
    long t = (long)blockIdx.x * blockDim.x + threadIdx.x;
    if (t >= E * 64) return;
    long e = t >> 6; int c4 = (int)(t & 63) * 4;
    long node = dsti[e];
    const float4 v = *(const float4*)(he + e * HID + c4);
    float* d = dstbuf + node * HID + c4;
    atomicAdd(d + 0, v.x);
    atomicAdd(d + 1, v.y);
    atomicAdd(d + 2, v.z);
    atomicAdd(d + 3, v.w);
}

// msg[e][:] = bf16( h_node[src[e]][:] - h_edge[e^1][:] )
__global__ void k_msg(__bf16* __restrict__ msg, const float* __restrict__ hn,
                      const float* __restrict__ he, const int* __restrict__ srci, long E)
{
    long t = (long)blockIdx.x * blockDim.x + threadIdx.x;
    if (t >= E * 64) return;
    long e = t >> 6; int c4 = (int)(t & 63) * 4;
    long s = srci[e];
    const float4 a = *(const float4*)(hn + s * HID + c4);
    const float4 b = *(const float4*)(he + (e ^ 1) * HID + c4);
    v4bf m; m.x = f2bf(a.x - b.x); m.y = f2bf(a.y - b.y);
            m.z = f2bf(a.z - b.z); m.w = f2bf(a.w - b.w);
    *(v4bf*)(msg + e * HID + c4) = m;
}

// A_lr[r, 0:256]=agg, [256:512]=h_node, [512:768]=h0_node  (bf16)
__global__ void k_concat3(__bf16* __restrict__ dst, const float* __restrict__ s0,
                          const float* __restrict__ s1, const float* __restrict__ s2,
                          long rows)
{
    long t = (long)blockIdx.x * blockDim.x + threadIdx.x;   // rows * 192 (768/4)
    if (t >= rows * 192) return;
    long r = t / 192; int c4 = (int)(t - r * 192) * 4;
    const float* s = (c4 < 256) ? s0 : (c4 < 512) ? s1 : s2;
    int cc = c4 & 255;
    const float4 v = *(const float4*)(s + r * HID + cc);
    v4bf m; m.x = f2bf(v.x); m.y = f2bf(v.y); m.z = f2bf(v.z); m.w = f2bf(v.w);
    *(v4bf*)(dst + r * 768 + c4) = m;
}

// A_o[r, 0:256]=h_node2, [256:512]=h0_node  (bf16)
__global__ void k_concat2(__bf16* __restrict__ dst, const float* __restrict__ s0,
                          const float* __restrict__ s1, long rows)
{
    long t = (long)blockIdx.x * blockDim.x + threadIdx.x;   // rows * 128 (512/4)
    if (t >= rows * 128) return;
    long r = t / 128; int c4 = (int)(t - r * 128) * 4;
    const float* s = (c4 < 256) ? s0 : s1;
    int cc = c4 & 255;
    const float4 v = *(const float4*)(s + r * HID + cc);
    v4bf m; m.x = f2bf(v.x); m.y = f2bf(v.y); m.z = f2bf(v.z); m.w = f2bf(v.w);
    *(v4bf*)(dst + r * 512 + c4) = m;
}

// ---------------------------------------------------------------------------
extern "C" void kernel_launch(void* const* d_in, const int* in_sizes, int n_in,
                              void* d_out, int out_size, void* d_ws, size_t ws_size,
                              hipStream_t stream)
{
    const float* node_attr = (const float*)d_in[0];
    const float* edge_attr = (const float*)d_in[1];
    const int*   src       = (const int*)d_in[2];
    const int*   dst       = (const int*)d_in[3];
    const float* W_i_atom  = (const float*)d_in[4];
    const float* W_i_bond  = (const float*)d_in[5];
    const float* W_h       = (const float*)d_in[6];
    const float* W_o       = (const float*)d_in[7];
    const float* W_o_b     = (const float*)d_in[8];
    const float* W_lr      = (const float*)d_in[9];
    float* out = (float*)d_out;

    char* ws = (char*)d_ws;
    const size_t SN = (size_t)NN * HID * sizeof(float);   // 102.4 MB
    const size_t SE = (size_t)NE * HID * sizeof(float);   // 409.6 MB
    float*  h0n  = (float*)(ws);
    float*  hn   = (float*)(ws + SN);
    float*  h0e  = (float*)(ws + 2 * SN);
    float*  he   = (float*)(ws + 2 * SN + SE);
    float*  agg  = (float*)(ws + 2 * SN + 2 * SE);
    __bf16* msgb = (__bf16*)(ws + 3 * SN + 2 * SE);                     // also A_lr
    __bf16* edA  = (__bf16*)(ws + 3 * SN + 2 * SE + SE / 2);            // also A_o
    __bf16* ndA  = (__bf16*)((char*)edA + (size_t)NE * KP_IN * 2);
    __bf16* wia  = (__bf16*)((char*)ndA + (size_t)NN * KP_IN * 2);
    __bf16* wib  = wia + HID * KP_IN;
    __bf16* wh   = wib + HID * KP_IN;         // 3 * 256 * 256
    __bf16* wlr  = wh + 3 * HID * HID;        // 256 * 768
    __bf16* wo   = wlr + HID * 768;           // 256 * 512

    const int B = 256;
    auto blocks = [](long n) { return (unsigned)((n + 255) / 256); };

    // weights -> transposed bf16
    k_wt_bf16<<<blocks(HID * KP_IN), B, 0, stream>>>(wia, W_i_atom, ATOM_F, KP_IN);
    k_wt_bf16<<<blocks(HID * KP_IN), B, 0, stream>>>(wib, W_i_bond, BOND_F, KP_IN);
    for (int d = 0; d < 3; ++d)
        k_wt_bf16<<<blocks(HID * HID), B, 0, stream>>>(wh + d * HID * HID,
                                                       W_h + (size_t)d * HID * HID, HID, HID);
    k_wt_bf16<<<blocks(HID * 768), B, 0, stream>>>(wlr, W_lr, 768, 768);
    k_wt_bf16<<<blocks(HID * 512), B, 0, stream>>>(wo, W_o, 512, 512);

    // features -> padded bf16
    k_pad_bf16<<<blocks((long)NN * KP_IN), B, 0, stream>>>(ndA, node_attr, NN, ATOM_F, KP_IN);
    k_pad_bf16<<<blocks((long)NE * KP_IN), B, 0, stream>>>(edA, edge_attr, NE, BOND_F, KP_IN);

    auto gemm = [&](float* o, const __bf16* A, const __bf16* Bt, int Kp, long M,
                    const float* addsrc, const float* bias, int relu) {
        const int use_lds = (Kp <= 256);
        const size_t smem = use_lds ? (size_t)HID * (Kp + LDS_PAD) * 2 : 0;
        k_gemm_bf16<<<(unsigned)(M / 32), 256, smem, stream>>>(
            o, A, Bt, addsrc, bias, M, Kp, relu, use_lds);
    };

    // input projections (+ ReLU)
    gemm(h0n, ndA, wia, KP_IN, NN, nullptr, nullptr, 1);
    gemm(h0e, edA, wib, KP_IN, NE, nullptr, nullptr, 1);
    k_copy_f32<<<blocks((long)NN * HID), B, 0, stream>>>(hn, h0n, (long)NN * HID);

    // message passing: 3 iterations
    for (int d = 0; d < 3; ++d) {
        const float* he_cur = (d == 0) ? h0e : he;
        k_scatter_add<<<blocks((long)NE * 64), B, 0, stream>>>(hn, he_cur, dst, NE);
        k_msg<<<blocks((long)NE * 64), B, 0, stream>>>(msgb, hn, he_cur, src, NE);
        gemm(he, msgb, wh + d * HID * HID, HID, NE, h0e, nullptr, 1);
    }

    // final aggregation
    k_fill0_f32<<<blocks((long)NN * HID), B, 0, stream>>>(agg, (long)NN * HID);
    k_scatter_add<<<blocks((long)NE * 64), B, 0, stream>>>(agg, he, dst, NE);

    // readout: concat(agg, h_node, h0_node) @ W_lr  (no relu)
    __bf16* A_lr = msgb;
    k_concat3<<<blocks((long)NN * 192), B, 0, stream>>>(A_lr, agg, hn, h0n, NN);
    float* hn2 = he;   // reuse edge buffer
    gemm(hn2, A_lr, wlr, 768, NN, nullptr, nullptr, 0);

    // out = relu(concat(h_node2, h0_node) @ W_o + b)
    __bf16* A_o = edA;
    k_concat2<<<blocks((long)NN * 128), B, 0, stream>>>(A_o, hn2, h0n, NN);
    gemm(out, A_o, wo, 512, NN, nullptr, W_o_b, 1);
}
